// CAM_71640054497378
// MI455X (gfx1250) — matrix operand order repaired and checked by reading, not compiled, not run on hardware
//
#include <hip/hip_runtime.h>
#include <hip/hip_bf16.h>
#include <math.h>

// CAM module for MI455X (gfx1250). All matmuls run on V_WMMA_F32_16X16X32_BF16
// (bf16 operands, fp32 accumulate), 2x2 tiles per wave, 128-bit fragment loads,
// compile-time specialized NT/NN paths.

typedef __bf16 bf16;
typedef bf16  bf16x16 __attribute__((ext_vector_type(16)));
typedef float f32x8   __attribute__((ext_vector_type(8)));
typedef unsigned short ushort8 __attribute__((ext_vector_type(8)));

#define NB   8      // batch
#define NC   64     // channels
#define NHW  1024   // H*W
#define ND   32     // C/2
#define NHWD 256    // (H/2)*(W/2)

union V16u { struct { ushort8 lo, hi; } p; bf16x16 v; };

// 16 contiguous bf16 (two b128 loads)
__device__ __forceinline__ bf16x16 load_b16x16(const bf16* p)
{
  V16u u;
  u.p.lo = *(const ushort8*)p;
  u.p.hi = *(const ushort8*)(p + 8);
  return u.v;
}
// A-fragment: two 8-bf16 chunks at p and p+16 (K-halves of the 16x32 tile)
__device__ __forceinline__ bf16x16 load_afrag(const bf16* p)
{
  V16u u;
  u.p.lo = *(const ushort8*)p;
  u.p.hi = *(const ushort8*)(p + 16);
  return u.v;
}

// ---------------------------------------------------------------------------
// Batched GEMM: C[b] = A[b] * opB(B[b]) (+bias_row +bias_col)
//   A is [M,K] row-major (always contiguous-K fragment loads)
//   opB(B) is [K,N]; TRANSB=1 => B stored [N,K] (NT, contiguous-K loads)
// One wave computes a 32x32 macro-tile: 2 A-frags x 2 B-frags -> 4 WMMAs.
// All branches wave-uniform => EXEC all ones at every WMMA.
// ---------------------------------------------------------------------------
template <int TRANSB>
__global__ void gemm_bf16_wmma(const bf16* __restrict__ A, const bf16* __restrict__ B,
                               void* __restrict__ C,
                               int M, int N, int K,
                               int lda, int ldb, int ldc,
                               long long sA, long long sB, long long sC,
                               const float* __restrict__ bias_row,
                               const float* __restrict__ bias_col,
                               int out_is_bf16)
{
  const int tilesN = N >> 5;
  const int tilesM = M >> 5;
  const int wave   = (int)(threadIdx.x >> 5);
  const int tile   = (int)blockIdx.x * 8 + wave;
  if (tile >= tilesM * tilesN) return;          // wave-uniform exit
  const int tm = tile / tilesN;
  const int tn = tile - tm * tilesN;
  const int lane = (int)(threadIdx.x & 31);
  const int hl   = lane >> 4;
  const int mr   = lane & 15;

  const int b = (int)blockIdx.y;
  const bf16* Ab = A + (size_t)b * (size_t)sA;
  const bf16* Bb = B + (size_t)b * (size_t)sB;
  const int r0 = tm * 32 + mr, r1 = r0 + 16;    // A-fragment rows
  const int c0 = tn * 32 + mr, c1 = c0 + 16;    // B-fragment cols

  // fragment pointers, bumped by constant strides in the loop
  const bf16* pa0 = Ab + (size_t)r0 * lda + (size_t)(hl * 16);
  const bf16* pa1 = Ab + (size_t)r1 * lda + (size_t)(hl * 16);
  const bf16* pb0;
  const bf16* pb1;
  if (TRANSB) {
    pb0 = Bb + (size_t)c0 * ldb + (size_t)(hl * 16);
    pb1 = Bb + (size_t)c1 * ldb + (size_t)(hl * 16);
  } else {
    pb0 = Bb + (size_t)(hl * 16) * ldb + c0;
    pb1 = pb0 + 16;                             // c1 = c0 + 16
  }

  f32x8 acc00 = {0.f,0.f,0.f,0.f,0.f,0.f,0.f,0.f};
  f32x8 acc01 = acc00, acc10 = acc00, acc11 = acc00;

#pragma unroll 2
  for (int k = 0; k < K; k += 32) {
    const bf16x16 a0 = load_b16x16(pa0); pa0 += 32;
    const bf16x16 a1 = load_b16x16(pa1); pa1 += 32;
    bf16x16 b0, b1;
    if (TRANSB) {
      b0 = load_b16x16(pb0); pb0 += 32;
      b1 = load_b16x16(pb1); pb1 += 32;
    } else {
#pragma unroll
      for (int j = 0; j < 16; ++j) {
        b0[j] = pb0[(size_t)j * ldb];
        b1[j] = pb1[(size_t)j * ldb];
      }
      pb0 += (size_t)32 * ldb;
      pb1 += (size_t)32 * ldb;
    }
    acc00 = __builtin_amdgcn_wmma_f32_16x16x32_bf16(false, a0, false, b0, (short)0, acc00, false, false);
    acc01 = __builtin_amdgcn_wmma_f32_16x16x32_bf16(false, a0, false, b1, (short)0, acc01, false, false);
    acc10 = __builtin_amdgcn_wmma_f32_16x16x32_bf16(false, a1, false, b0, (short)0, acc10, false, false);
    acc11 = __builtin_amdgcn_wmma_f32_16x16x32_bf16(false, a1, false, b1, (short)0, acc11, false, false);
  }

  const size_t cb = (size_t)b * (size_t)sC;
#pragma unroll
  for (int r = 0; r < 8; ++r) {
    const int m0r = tm * 32 + r + hl * 8;       // rows of acc0x
    const int m1r = m0r + 16;                   // rows of acc1x
    float v00 = acc00[r], v01 = acc01[r], v10 = acc10[r], v11 = acc11[r];
    if (bias_row) { v00 += bias_row[m0r]; v01 += bias_row[m0r]; v10 += bias_row[m1r]; v11 += bias_row[m1r]; }
    if (bias_col) { v00 += bias_col[c0];  v01 += bias_col[c1];  v10 += bias_col[c0];  v11 += bias_col[c1]; }
    if (out_is_bf16) {
      bf16* O = (bf16*)C;
      O[cb + (size_t)m0r * ldc + c0] = (bf16)v00;
      O[cb + (size_t)m0r * ldc + c1] = (bf16)v01;
      O[cb + (size_t)m1r * ldc + c0] = (bf16)v10;
      O[cb + (size_t)m1r * ldc + c1] = (bf16)v11;
    } else {
      float* O = (float*)C;
      O[cb + (size_t)m0r * ldc + c0] = v00;
      O[cb + (size_t)m0r * ldc + c1] = v01;
      O[cb + (size_t)m1r * ldc + c0] = v10;
      O[cb + (size_t)m1r * ldc + c1] = v11;
    }
  }
}

// ---------------------------------------------------------------------------
// fp32 -> bf16 conversion
// ---------------------------------------------------------------------------
__global__ void cvt_f32_bf16(const float* __restrict__ in, bf16* __restrict__ out, int n)
{
  int i = (int)(blockIdx.x * blockDim.x + threadIdx.x);
  if (i < n) out[i] = (bf16)in[i];
}

// LDS-tiled transpose + convert: in [b, M, N] f32 -> out [b, N, M] bf16
__global__ void transpose_cvt_bf16(const float* __restrict__ in, bf16* __restrict__ out,
                                   int M, int N)
{
  __shared__ float t[16][17];
  const int b = (int)blockIdx.z;
  const int n0 = (int)blockIdx.x * 16, m0 = (int)blockIdx.y * 16;
  const float* ib = in + (size_t)b * M * N;
  bf16* ob = out + (size_t)b * M * N;
  t[threadIdx.y][threadIdx.x] = ib[(size_t)(m0 + threadIdx.y) * N + n0 + threadIdx.x];
  __syncthreads();
  ob[(size_t)(n0 + threadIdx.y) * M + m0 + threadIdx.x] = (bf16)t[threadIdx.x][threadIdx.y];
}

// ---------------------------------------------------------------------------
// Row softmax: fp32 logits in, bf16 probabilities out. One block per row.
// ---------------------------------------------------------------------------
__global__ void softmax_rows(const float* __restrict__ in, bf16* __restrict__ out, int N)
{
  const size_t row = blockIdx.x;
  const float* p = in + row * (size_t)N;
  bf16* q = out + row * (size_t)N;
  __shared__ float red[256];
  float m = -1e30f;
  for (int i = threadIdx.x; i < N; i += 256) m = fmaxf(m, p[i]);
  red[threadIdx.x] = m; __syncthreads();
  for (int s = 128; s > 0; s >>= 1) {
    if ((int)threadIdx.x < s) red[threadIdx.x] = fmaxf(red[threadIdx.x], red[threadIdx.x + s]);
    __syncthreads();
  }
  m = red[0]; __syncthreads();
  float sum = 0.f;
  for (int i = threadIdx.x; i < N; i += 256) sum += __expf(p[i] - m);
  red[threadIdx.x] = sum; __syncthreads();
  for (int s = 128; s > 0; s >>= 1) {
    if ((int)threadIdx.x < s) red[threadIdx.x] += red[threadIdx.x + s];
    __syncthreads();
  }
  const float inv = 1.0f / red[0];
  for (int i = threadIdx.x; i < N; i += 256) q[i] = (bf16)(__expf(p[i] - m) * inv);
}

// ---------------------------------------------------------------------------
// Bicubic helpers (Keys a=-0.5, matching jax.image.resize 'bicubic')
// ---------------------------------------------------------------------------
__device__ __forceinline__ float cubic_w(float t)
{
  const float a = -0.5f;
  float at = fabsf(t);
  if (at <= 1.f) return ((a + 2.f) * at - (a + 3.f)) * at * at + 1.f;
  if (at < 2.f)  return ((((at - 5.f) * at) + 8.f) * at - 4.f) * a;
  return 0.f;
}

// down_feat: antialiased bicubic 32x32 -> 16x16, output written TRANSPOSED as
// [B, 256, 64] bf16 so the downsampled feature gram is a contiguous NT GEMM.
__global__ void down_bicubic2_T(const float* __restrict__ x, bf16* __restrict__ outT, int BC)
{
  int idx = (int)(blockIdx.x * blockDim.x + threadIdx.x);
  if (idx >= BC * 256) return;
  int bc = idx >> 8, r = idx & 255;
  int oy = r >> 4, ox = r & 15;
  float cy = (oy + 0.5f) * 2.f - 0.5f;
  float cx = (ox + 0.5f) * 2.f - 0.5f;
  int y0 = (int)floorf(cy) - 3, x0 = (int)floorf(cx) - 3;
  const float* img = x + (size_t)bc * 1024;
  float acc = 0.f, wsum = 0.f;
  for (int dy = 0; dy < 8; ++dy) {
    int yy = y0 + dy;
    float wy = cubic_w((yy - cy) * 0.5f);
    int yc = yy < 0 ? 0 : (yy > 31 ? 31 : yy);
    for (int dx = 0; dx < 8; ++dx) {
      int xx = x0 + dx;
      float wx = cubic_w((xx - cx) * 0.5f);
      int xc = xx < 0 ? 0 : (xx > 31 ? 31 : xx);
      float w = wy * wx;
      acc += w * img[yc * 32 + xc];
      wsum += w;
    }
  }
  int b = bc >> 6, c = bc & 63;
  outT[(size_t)b * (NHWD * NC) + (size_t)r * NC + c] = (bf16)(acc / wsum);
}

// up_feat: bicubic 4x upsample of [B,256,256] -> [B,1024,1024], written
// TRANSPOSED (outT[b][q][p] = U[b][p][q]) => consuming GEMM uses NT path.
__global__ void up4_bicubic_T(const bf16* __restrict__ s, bf16* __restrict__ outT, int batch)
{
  size_t idx = (size_t)blockIdx.x * blockDim.x + threadIdx.x;
  if (idx >= (size_t)batch * 1024 * 1024) return;
  int b = (int)(idx >> 20);
  int rem = (int)(idx & 1048575);
  int p = rem >> 10, q = rem & 1023;
  float sp = (p + 0.5f) * 0.25f - 0.5f;
  float sq = (q + 0.5f) * 0.25f - 0.5f;
  int py = (int)floorf(sp), qx = (int)floorf(sq);
  const bf16* img = s + (size_t)b * 65536;
  float acc = 0.f, wsum = 0.f;
  for (int dy = -1; dy <= 2; ++dy) {
    float wy = cubic_w(sp - (float)(py + dy));
    int yc = py + dy; yc = yc < 0 ? 0 : (yc > 255 ? 255 : yc);
    for (int dx = -1; dx <= 2; ++dx) {
      float wx = cubic_w(sq - (float)(qx + dx));
      int xc = qx + dx; xc = xc < 0 ? 0 : (xc > 255 ? 255 : xc);
      float w = wy * wx;
      acc += w * (float)img[yc * 256 + xc];
      wsum += w;
    }
  }
  outT[(size_t)b * 1048576 + (size_t)q * 1024 + p] = (bf16)(acc / wsum);
}

// ---------------------------------------------------------------------------
// BatchNorm batch statistics: per (tensor, channel) mean and rstd over B*HW.
// blockIdx.x = tensor*64 + channel; tensors: 0=ch1o 1=ch2o 2=hw1o 3=hw2o
// ---------------------------------------------------------------------------
__global__ void bn_stats(const float* __restrict__ t0, const float* __restrict__ t1,
                         const float* __restrict__ t2, const float* __restrict__ t3,
                         float* __restrict__ stats)
{
  int tz = (int)blockIdx.x >> 6, c = (int)blockIdx.x & 63;
  const float* t = (tz == 0) ? t0 : (tz == 1) ? t1 : (tz == 2) ? t2 : t3;
  __shared__ float s1[256], s2[256];
  float a = 0.f, b = 0.f;
  for (int i = threadIdx.x; i < NB * NHW; i += 256) {
    int bb = i >> 10, p = i & 1023;
    float v = t[((size_t)(bb * NC + c)) * NHW + p];
    a += v; b += v * v;
  }
  s1[threadIdx.x] = a; s2[threadIdx.x] = b; __syncthreads();
  for (int s = 128; s > 0; s >>= 1) {
    if ((int)threadIdx.x < s) { s1[threadIdx.x] += s1[threadIdx.x + s]; s2[threadIdx.x] += s2[threadIdx.x + s]; }
    __syncthreads();
  }
  if (threadIdx.x == 0) {
    float mean = s1[0] * (1.f / 8192.f);
    float var  = s2[0] * (1.f / 8192.f) - mean * mean;
    stats[blockIdx.x * 2]     = mean;
    stats[blockIdx.x * 2 + 1] = rsqrtf(var + 1e-5f);
  }
}

// Fused BN + residual + ReLU + average of the two paths, both outputs.
__global__ void fuse_out(const float* __restrict__ ch1, const float* __restrict__ ch2,
                         const float* __restrict__ hw1, const float* __restrict__ hw2,
                         const float* __restrict__ x1,  const float* __restrict__ x2,
                         const float* __restrict__ gamma, const float* __restrict__ beta,
                         const float* __restrict__ stats, float* __restrict__ out)
{
  int idx = (int)(blockIdx.x * blockDim.x + threadIdx.x);
  if (idx >= NB * NC * NHW) return;
  int c = (idx >> 10) & 63;
  float g = gamma[c], be = beta[c];
  float m0 = stats[(0 * 64 + c) * 2], r0 = stats[(0 * 64 + c) * 2 + 1];
  float m1 = stats[(1 * 64 + c) * 2], r1 = stats[(1 * 64 + c) * 2 + 1];
  float m2 = stats[(2 * 64 + c) * 2], r2 = stats[(2 * 64 + c) * 2 + 1];
  float m3 = stats[(3 * 64 + c) * 2], r3 = stats[(3 * 64 + c) * 2 + 1];
  float v1 = x1[idx], v2 = x2[idx];
  float a1 = fmaxf((ch1[idx] - m0) * r0 * g + be + v1, 0.f);
  float h1 = fmaxf((hw1[idx] - m2) * r2 * g + be + v1, 0.f);
  float a2 = fmaxf((ch2[idx] - m1) * r1 * g + be + v2, 0.f);
  float h2 = fmaxf((hw2[idx] - m3) * r3 * g + be + v2, 0.f);
  out[idx]                 = 0.5f * (a1 + h1);
  out[NB * NC * NHW + idx] = 0.5f * (a2 + h2);
}

// ---------------------------------------------------------------------------
// Host orchestration
// ---------------------------------------------------------------------------
static inline void run_gemm(hipStream_t st, const void* A, const void* B, void* C,
                            int M, int N, int K, int lda, int ldb, int ldc,
                            long long sA, long long sB, long long sC,
                            int tB, const float* br, const float* bc,
                            int obf, int batch)
{
  int tiles = (M / 32) * (N / 32);
  dim3 grid((unsigned)((tiles + 7) / 8), (unsigned)batch);
  if (tB)
    gemm_bf16_wmma<1><<<grid, 256, 0, st>>>((const bf16*)A, (const bf16*)B, C, M, N, K,
                                            lda, ldb, ldc, sA, sB, sC, br, bc, obf);
  else
    gemm_bf16_wmma<0><<<grid, 256, 0, st>>>((const bf16*)A, (const bf16*)B, C, M, N, K,
                                            lda, ldb, ldc, sA, sB, sC, br, bc, obf);
}

extern "C" void kernel_launch(void* const* d_in, const int* in_sizes, int n_in,
                              void* d_out, int out_size, void* d_ws, size_t ws_size,
                              hipStream_t stream)
{
  (void)in_sizes; (void)n_in; (void)out_size; (void)ws_size;
  const float* x1     = (const float*)d_in[0];
  const float* x2     = (const float*)d_in[1];
  const float* w_down = (const float*)d_in[2];
  const float* b_down = (const float*)d_in[3];
  const float* w_up   = (const float*)d_in[4];
  const float* b_up   = (const float*)d_in[5];
  const float* gamma  = (const float*)d_in[6];
  const float* beta   = (const float*)d_in[7];
  float* out = (float*)d_out;

  size_t off = 0;
  auto alloc = [&](size_t bytes) -> void* {
    void* p = (char*)d_ws + off;
    off += (bytes + 255) & ~(size_t)255;
    return p;
  };

  // bf16 operand buffers
  bf16* Xb1 = (bf16*)alloc((size_t)NB * NC * NHW * 2);   // [B,C,HW]
  bf16* Xb2 = (bf16*)alloc((size_t)NB * NC * NHW * 2);
  bf16* XT1 = (bf16*)alloc((size_t)NB * NHW * NC * 2);   // [B,HW,C]  (X^T)
  bf16* XT2 = (bf16*)alloc((size_t)NB * NHW * NC * 2);
  bf16* Wdb = (bf16*)alloc((size_t)ND * NC * 2);
  bf16* Wub = (bf16*)alloc((size_t)NC * ND * 2);
  bf16* Y1  = (bf16*)alloc((size_t)NB * ND * NHW * 2);
  bf16* Y2  = (bf16*)alloc((size_t)NB * ND * NHW * 2);
  bf16* Gc1 = (bf16*)alloc((size_t)NB * NC * NC * 2);
  bf16* Gc2 = (bf16*)alloc((size_t)NB * NC * NC * 2);
  bf16* Gd1 = (bf16*)alloc((size_t)NB * ND * ND * 2);
  bf16* Gd2 = (bf16*)alloc((size_t)NB * ND * ND * 2);
  float* Lc = (float*)alloc((size_t)NB * NC * NC * 4);
  float* Ld = (float*)alloc((size_t)NB * ND * ND * 4);
  bf16* Sc1 = (bf16*)alloc((size_t)NB * NC * NC * 2);
  bf16* Sc2 = (bf16*)alloc((size_t)NB * NC * NC * 2);
  bf16* Sd1 = (bf16*)alloc((size_t)NB * ND * ND * 2);
  bf16* Sd2 = (bf16*)alloc((size_t)NB * ND * ND * 2);
  bf16* T1  = (bf16*)alloc((size_t)NB * NC * ND * 2);
  bf16* T2  = (bf16*)alloc((size_t)NB * NC * ND * 2);
  bf16* Uc1 = (bf16*)alloc((size_t)NB * NC * NC * 2);
  bf16* Uc2 = (bf16*)alloc((size_t)NB * NC * NC * 2);
  bf16* CH1 = (bf16*)alloc((size_t)NB * NC * NC * 2);
  bf16* CH2 = (bf16*)alloc((size_t)NB * NC * NC * 2);
  bf16* XdT1 = (bf16*)alloc((size_t)NB * NHWD * NC * 2); // [B,256,64] (Xd^T)
  bf16* XdT2 = (bf16*)alloc((size_t)NB * NHWD * NC * 2);
  bf16* Fd1 = (bf16*)alloc((size_t)NB * NHWD * NHWD * 2);
  bf16* Fd2 = (bf16*)alloc((size_t)NB * NHWD * NHWD * 2);
  float* Lfd = (float*)alloc((size_t)NB * NHWD * NHWD * 4);
  bf16* Sfd1 = (bf16*)alloc((size_t)NB * NHWD * NHWD * 2);
  bf16* Sfd2 = (bf16*)alloc((size_t)NB * NHWD * NHWD * 2);
  float* cho1 = (float*)alloc((size_t)NB * NC * NHW * 4);
  float* cho2 = (float*)alloc((size_t)NB * NC * NHW * 4);
  float* hwo1 = (float*)alloc((size_t)NB * NC * NHW * 4);
  float* hwo2 = (float*)alloc((size_t)NB * NC * NHW * 4);
  float* stats = (float*)alloc(4 * 64 * 2 * 4);
  // big [B,1024,1024] bf16 buffers (16MB each), with reuse
  const size_t BIGE = (size_t)NB * NHW * NHW;
  bf16* BIG0 = (bf16*)alloc(BIGE * 2);   // F1, then U1T
  bf16* BIG1 = (bf16*)alloc(BIGE * 2);   // F2, then U2T
  bf16* BIG2 = (bf16*)alloc(BIGE * 2);   // S1, then HW2
  bf16* BIG3 = (bf16*)alloc(BIGE * 2);   // S2
  bf16* BIG4 = (bf16*)alloc(BIGE * 2);   // HW1
  float* LF  = (float*)alloc(BIGE * 4);  // shared fp32 logits buffer

  const int TPB = 256;
  const long long sX  = (long long)NC * NHW;     // 65536
  const long long sXT = (long long)NHW * NC;     // 65536
  const long long sY  = (long long)ND * NHW;     // 32768
  const long long sGc = (long long)NC * NC;      // 4096
  const long long sGd = (long long)ND * ND;      // 1024
  const long long sFd = (long long)NHWD * NHWD;  // 65536
  const long long sF  = (long long)NHW * NHW;    // 1048576

  // --- operand conversion / transposition to bf16 ---
  cvt_f32_bf16<<<(NB * NC * NHW + TPB - 1) / TPB, TPB, 0, stream>>>(x1, Xb1, NB * NC * NHW);
  cvt_f32_bf16<<<(NB * NC * NHW + TPB - 1) / TPB, TPB, 0, stream>>>(x2, Xb2, NB * NC * NHW);
  cvt_f32_bf16<<<(ND * NC + TPB - 1) / TPB, TPB, 0, stream>>>(w_down, Wdb, ND * NC);
  cvt_f32_bf16<<<(NC * ND + TPB - 1) / TPB, TPB, 0, stream>>>(w_up, Wub, NC * ND);
  transpose_cvt_bf16<<<dim3(NHW / 16, NC / 16, NB), dim3(16, 16), 0, stream>>>(x1, XT1, NC, NHW);
  transpose_cvt_bf16<<<dim3(NHW / 16, NC / 16, NB), dim3(16, 16), 0, stream>>>(x2, XT2, NC, NHW);

  // --- channel path ---
  // Y = w_down @ X + b_down == NT(Wdb, X^T)   [B,32,1024]
  run_gemm(stream, Wdb, XT1, Y1, ND, NHW, NC, NC, NC, NHW, 0, sXT, sY, 1, b_down, nullptr, 1, NB);
  run_gemm(stream, Wdb, XT2, Y2, ND, NHW, NC, NC, NC, NHW, 0, sXT, sY, 1, b_down, nullptr, 1, NB);
  // channel grams G = X X^T (NT)
  run_gemm(stream, Xb1, Xb1, Gc1, NC, NC, NHW, NHW, NHW, NC, sX, sX, sGc, 1, nullptr, nullptr, 1, NB);
  run_gemm(stream, Xb2, Xb2, Gc2, NC, NC, NHW, NHW, NHW, NC, sX, sX, sGc, 1, nullptr, nullptr, 1, NB);
  run_gemm(stream, Y1, Y1, Gd1, ND, ND, NHW, NHW, NHW, ND, sY, sY, sGd, 1, nullptr, nullptr, 1, NB);
  run_gemm(stream, Y2, Y2, Gd2, ND, ND, NHW, NHW, NHW, ND, sY, sY, sGd, 1, nullptr, nullptr, 1, NB);
  // channel attentions
  run_gemm(stream, Gc1, Gc2, Lc, NC, NC, NC, NC, NC, NC, sGc, sGc, sGc, 1, nullptr, nullptr, 0, NB);
  softmax_rows<<<NB * NC, 256, 0, stream>>>(Lc, Sc1, NC);
  run_gemm(stream, Gc2, Gc1, Lc, NC, NC, NC, NC, NC, NC, sGc, sGc, sGc, 1, nullptr, nullptr, 0, NB);
  softmax_rows<<<NB * NC, 256, 0, stream>>>(Lc, Sc2, NC);
  run_gemm(stream, Gd1, Gd2, Ld, ND, ND, ND, ND, ND, ND, sGd, sGd, sGd, 1, nullptr, nullptr, 0, NB);
  softmax_rows<<<NB * ND, 256, 0, stream>>>(Ld, Sd1, ND);
  run_gemm(stream, Gd2, Gd1, Ld, ND, ND, ND, ND, ND, ND, sGd, sGd, sGd, 1, nullptr, nullptr, 0, NB);
  softmax_rows<<<NB * ND, 256, 0, stream>>>(Ld, Sd2, ND);
  // up_channel: T = w_up @ S + b_up (rows, NN);  U = T w_up^T + b_up (cols, NT)
  run_gemm(stream, Wub, Sd1, T1, NC, ND, ND, ND, ND, ND, 0, sGd, (long long)NC * ND, 0, b_up, nullptr, 1, NB);
  run_gemm(stream, Wub, Sd2, T2, NC, ND, ND, ND, ND, ND, 0, sGd, (long long)NC * ND, 0, b_up, nullptr, 1, NB);
  run_gemm(stream, T1, Wub, Uc1, NC, NC, ND, ND, ND, NC, (long long)NC * ND, 0, sGc, 1, nullptr, b_up, 1, NB);
  run_gemm(stream, T2, Wub, Uc2, NC, NC, ND, ND, ND, NC, (long long)NC * ND, 0, sGc, 1, nullptr, b_up, 1, NB);
  // ch = attn_c @ U (NN, tiny) ; ch_o = ch @ X == NT(ch, X^T)
  run_gemm(stream, Sc1, Uc1, CH1, NC, NC, NC, NC, NC, NC, sGc, sGc, sGc, 0, nullptr, nullptr, 1, NB);
  run_gemm(stream, Sc2, Uc2, CH2, NC, NC, NC, NC, NC, NC, sGc, sGc, sGc, 0, nullptr, nullptr, 1, NB);
  run_gemm(stream, CH1, XT1, cho1, NC, NHW, NC, NC, NC, NHW, sGc, sXT, sX, 1, nullptr, nullptr, 0, NB);
  run_gemm(stream, CH2, XT2, cho2, NC, NHW, NC, NC, NC, NHW, sGc, sXT, sX, 1, nullptr, nullptr, 0, NB);

  // --- feature path ---
  down_bicubic2_T<<<(NB * NC * 256 + TPB - 1) / TPB, TPB, 0, stream>>>(x1, XdT1, NB * NC);
  down_bicubic2_T<<<(NB * NC * 256 + TPB - 1) / TPB, TPB, 0, stream>>>(x2, XdT2, NB * NC);
  // feature grams: F = X^T X == NT(X^T, X^T), K = 64, all-contiguous loads
  run_gemm(stream, XT1, XT1, BIG0, NHW, NHW, NC, NC, NC, NHW, sXT, sXT, sF, 1, nullptr, nullptr, 1, NB);
  run_gemm(stream, XT2, XT2, BIG1, NHW, NHW, NC, NC, NC, NHW, sXT, sXT, sF, 1, nullptr, nullptr, 1, NB);
  run_gemm(stream, XdT1, XdT1, Fd1, NHWD, NHWD, NC, NC, NC, NHWD, (long long)NHWD * NC, (long long)NHWD * NC, sFd, 1, nullptr, nullptr, 1, NB);
  run_gemm(stream, XdT2, XdT2, Fd2, NHWD, NHWD, NC, NC, NC, NHWD, (long long)NHWD * NC, (long long)NHWD * NC, sFd, 1, nullptr, nullptr, 1, NB);
  // downsampled feature attentions
  run_gemm(stream, Fd1, Fd2, Lfd, NHWD, NHWD, NHWD, NHWD, NHWD, NHWD, sFd, sFd, sFd, 1, nullptr, nullptr, 0, NB);
  softmax_rows<<<NB * NHWD, 256, 0, stream>>>(Lfd, Sfd1, NHWD);
  run_gemm(stream, Fd2, Fd1, Lfd, NHWD, NHWD, NHWD, NHWD, NHWD, NHWD, sFd, sFd, sFd, 1, nullptr, nullptr, 0, NB);
  softmax_rows<<<NB * NHWD, 256, 0, stream>>>(Lfd, Sfd2, NHWD);
  // full-res feature attentions (the 17-GFLOP GEMMs)
  run_gemm(stream, BIG0, BIG1, LF, NHW, NHW, NHW, NHW, NHW, NHW, sF, sF, sF, 1, nullptr, nullptr, 0, NB);
  softmax_rows<<<NB * NHW, 256, 0, stream>>>(LF, BIG2, NHW);   // S1
  run_gemm(stream, BIG1, BIG0, LF, NHW, NHW, NHW, NHW, NHW, NHW, sF, sF, sF, 1, nullptr, nullptr, 0, NB);
  softmax_rows<<<NB * NHW, 256, 0, stream>>>(LF, BIG3, NHW);   // S2
  // 4x bicubic upsample, transposed output => NT GEMM next (F buffers now free)
  up4_bicubic_T<<<(unsigned)((BIGE + TPB - 1) / TPB), TPB, 0, stream>>>(Sfd1, BIG0, NB); // U1^T
  up4_bicubic_T<<<(unsigned)((BIGE + TPB - 1) / TPB), TPB, 0, stream>>>(Sfd2, BIG1, NB); // U2^T
  // hw = S @ U == NT(S, U^T)
  run_gemm(stream, BIG2, BIG0, BIG4, NHW, NHW, NHW, NHW, NHW, NHW, sF, sF, sF, 1, nullptr, nullptr, 1, NB); // HW1
  run_gemm(stream, BIG3, BIG1, BIG2, NHW, NHW, NHW, NHW, NHW, NHW, sF, sF, sF, 1, nullptr, nullptr, 1, NB); // HW2
  // hw_o[c][p] = sum_q X[c][q] * hw[p][q] == NT(X, HW)
  run_gemm(stream, Xb1, BIG4, hwo1, NC, NHW, NHW, NHW, NHW, NHW, sX, sF, sX, 1, nullptr, nullptr, 0, NB);
  run_gemm(stream, Xb2, BIG2, hwo2, NC, NHW, NHW, NHW, NHW, NHW, sX, sF, sX, 1, nullptr, nullptr, 0, NB);

  // --- epilogue: BN(batch stats) + residual + ReLU + 0.5*(ch+hw) ---
  bn_stats<<<4 * 64, 256, 0, stream>>>(cho1, cho2, hwo1, hwo2, stats);
  fuse_out<<<(NB * NC * NHW + TPB - 1) / TPB, TPB, 0, stream>>>(cho1, cho2, hwo1, hwo2,
                                                                x1, x2, gamma, beta, stats, out);
}
